// CausalSelfAttention_15075335209283
// MI455X (gfx1250) — matrix operand order repaired and checked by reading, not compiled, not run on hardware
//
#include <hip/hip_runtime.h>
#include <hip/hip_bf16.h>
#include <stdint.h>
#include <stddef.h>

#define T_SEQ 4096
#define C_DIM 768
#define H_NUM 12
#define HD    64
#define C3    (3 * C_DIM)

typedef __attribute__((ext_vector_type(16))) __bf16 bf16x16;
typedef __attribute__((ext_vector_type(8)))  float  f32x8;

union FragU { bf16x16 v; uint32_t u[8]; };

__device__ __forceinline__ f32x8 fzero() {
  f32x8 z;
#pragma unroll
  for (int i = 0; i < 8; ++i) z[i] = 0.0f;
  return z;
}

// A-fragment 16x32 bf16 from a row-major [M][K] source (ld = row stride in elems).
// Lanes 0-15: row = lane, K = {0..7, 16..23}; lanes 16-31: K = {8..15, 24..31}.
__device__ __forceinline__ bf16x16 load_fragA(const __bf16* __restrict__ src, int ld, int lane) {
  const int half = lane >> 4;
  const int r    = lane & 15;
  const __bf16* base = src + (size_t)r * ld + half * 8;
  FragU f;
#pragma unroll
  for (int i = 0; i < 4; ++i) {
    f.u[i]     = *reinterpret_cast<const uint32_t*>(base + 2 * i);
    f.u[4 + i] = *reinterpret_cast<const uint32_t*>(base + 16 + 2 * i);
  }
  return f.v;
}

// B-fragment 32x16 bf16. Source must be "transposed" row-major [N][K].
// Lanes 0-15: col = lane, K = {0..7, 16..23}; lanes 16-31: K = {8..15, 24..31}.
__device__ __forceinline__ bf16x16 load_fragB(const __bf16* __restrict__ src, int ld, int lane) {
  const int half = lane >> 4;
  const int n    = lane & 15;
  const __bf16* base = src + (size_t)n * ld + half * 8;
  FragU f;
#pragma unroll
  for (int i = 0; i < 4; ++i) {
    f.u[i]     = *reinterpret_cast<const uint32_t*>(base + 2 * i);
    f.u[4 + i] = *reinterpret_cast<const uint32_t*>(base + 16 + 2 * i);
  }
  return f.v;
}

__device__ __forceinline__ f32x8 wmma_bf16(bf16x16 a, bf16x16 b, f32x8 c) {
  return __builtin_amdgcn_wmma_f32_16x16x32_bf16(false, a, false, b, (short)0, c, false, false);
}

// ---------------- conversion kernels ----------------

__global__ __launch_bounds__(256) void cvt_f32_bf16(const float* __restrict__ in,
                                                    __bf16* __restrict__ out, int n) {
  int i = blockIdx.x * blockDim.x + threadIdx.x;
  if (i < n) out[i] = (__bf16)in[i];
}

// out[c*R + r] = in[r*Ccols + c]  (transpose + convert); out has R*Ccols elems.
__global__ __launch_bounds__(256) void cvt_transpose_bf16(const float* __restrict__ in,
                                                          __bf16* __restrict__ out,
                                                          int R, int Ccols) {
  int o = blockIdx.x * blockDim.x + threadIdx.x;
  if (o >= R * Ccols) return;
  int r = o % R;
  int c = o / R;
  out[o] = (__bf16)in[(size_t)r * Ccols + c];
}

// ---------------- QKV GEMM: 32x64 tile per wave, scattered per-head epilogue ----------------

__global__ __launch_bounds__(256) void qkv_gemm_kernel(const __bf16* __restrict__ xb,
                                                       const __bf16* __restrict__ waT,
                                                       __bf16* __restrict__ qo,
                                                       __bf16* __restrict__ ko,
                                                       __bf16* __restrict__ vTo) {
  const int lane = threadIdx.x & 31;
  const int wave = blockIdx.x * 8 + (threadIdx.x >> 5);
  const int tiles_n = C3 / 64;                       // 36
  if (wave >= (T_SEQ / 32) * tiles_n) return;
  const int tm = wave / tiles_n;
  const int tn = wave % tiles_n;

  const __bf16* a0 = xb  + (size_t)(tm * 32)      * C_DIM;
  const __bf16* a1 = xb  + (size_t)(tm * 32 + 16) * C_DIM;
  const __bf16* bt = waT + (size_t)(tn * 64)      * C_DIM;

  f32x8 acc[2][4];
#pragma unroll
  for (int i = 0; i < 2; ++i)
#pragma unroll
    for (int j = 0; j < 4; ++j) acc[i][j] = fzero();

#pragma unroll 2
  for (int kk = 0; kk < C_DIM; kk += 32) {
    const bf16x16 A0 = load_fragA(a0 + kk, C_DIM, lane);
    const bf16x16 A1 = load_fragA(a1 + kk, C_DIM, lane);
#pragma unroll
    for (int j = 0; j < 4; ++j) {
      const bf16x16 B = load_fragB(bt + (size_t)(j * 16) * C_DIM + kk, C_DIM, lane);
      acc[0][j] = wmma_bf16(A0, B, acc[0][j]);
      acc[1][j] = wmma_bf16(A1, B, acc[1][j]);
    }
  }

  const int half = lane >> 4;
  const int n    = lane & 15;
  const int col0  = tn * 64;                 // 64-col tile == exactly one head slice
  const int which = col0 / C_DIM;            // 0=q 1=k 2=v
  const int h     = (col0 % C_DIM) / HD;

#pragma unroll
  for (int i = 0; i < 2; ++i) {
#pragma unroll
    for (int r = 0; r < 8; ++r) {
      const int row = tm * 32 + i * 16 + half * 8 + r;
#pragma unroll
      for (int j = 0; j < 4; ++j) {
        const int d = j * 16 + n;
        const __bf16 val = (__bf16)acc[i][j][r];
        if (which == 0)      qo[(size_t)(h * T_SEQ + row) * HD + d] = val;
        else if (which == 1) ko[(size_t)(h * T_SEQ + row) * HD + d] = val;
        else                 vTo[(size_t)(h * HD + d) * T_SEQ + row] = val;  // V transposed
      }
    }
  }
}

// ---------------- causal flash attention: one wave per (head, 32 q-rows) ----------------

__global__ __launch_bounds__(256) void attn_kernel(const __bf16* __restrict__ q,
                                                   const __bf16* __restrict__ k,
                                                   const __bf16* __restrict__ vT,
                                                   __bf16* __restrict__ yb) {
  __shared__ __bf16 pbuf[8][32 * 32];                // per-wave P tiles (2 x 16x32 bf16)
  const int lane  = threadIdx.x & 31;
  const int wslot = threadIdx.x >> 5;
  const int wave  = blockIdx.x * 8 + wslot;
  const int qtiles = T_SEQ / 32;
  const int h  = wave / qtiles;
  const int qt = wave % qtiles;
  if (h >= H_NUM) return;

  const int half = lane >> 4;
  const int n    = lane & 15;

  // Q fragments: 2 row-tiles x 2 d-chunks
  bf16x16 qa[2][2];
#pragma unroll
  for (int i = 0; i < 2; ++i) {
    const __bf16* qbase = q + (size_t)(h * T_SEQ + qt * 32 + i * 16) * HD;
    qa[i][0] = load_fragA(qbase,      HD, lane);
    qa[i][1] = load_fragA(qbase + 32, HD, lane);
  }

  f32x8 acc[2][4];
  float mrow[2][8], lrow[2][8];
#pragma unroll
  for (int i = 0; i < 2; ++i) {
#pragma unroll
    for (int j = 0; j < 4; ++j) acc[i][j] = fzero();
#pragma unroll
    for (int r = 0; r < 8; ++r) { mrow[i][r] = -1e30f; lrow[i][r] = 0.0f; }
  }

  const float scale = 0.125f;                 // 1/sqrt(64)
  const int qrow0[2] = { qt * 32 + half * 8, qt * 32 + 16 + half * 8 };
  const __bf16* khead = k  + (size_t)h * T_SEQ * HD;
  const __bf16* vhead = vT + (size_t)(h * HD) * T_SEQ;

  for (int kb = 0; kb <= qt; ++kb) {          // key blocks of 32; kbmax == qt for both tiles
    const __bf16* kbase = khead + (size_t)(kb * 32) * HD;
    if (kb < qt) {                            // warm next K block into caches
      __builtin_prefetch(kbase + 32 * HD + (size_t)n * 2 * HD, 0, 1);
    }

    // Shared K fragments: 2 key sub-tiles x 2 d-chunks
    const bf16x16 kf[2][2] = {
      { load_fragB(kbase,                HD, lane), load_fragB(kbase + 32,           HD, lane) },
      { load_fragB(kbase + 16 * HD,      HD, lane), load_fragB(kbase + 16 * HD + 32, HD, lane) }
    };

    // ---- S = Q K^T for both 16-row tiles ----
    f32x8 s[2][2];
#pragma unroll
    for (int i = 0; i < 2; ++i) {
#pragma unroll
      for (int t = 0; t < 2; ++t) {
        f32x8 sv = fzero();
        sv = wmma_bf16(qa[i][0], kf[t][0], sv);
        sv = wmma_bf16(qa[i][1], kf[t][1], sv);
        s[i][t] = sv;
      }
    }

    // ---- scale + causal mask + online softmax + P into LDS ----
    const int kcol0 = kb * 32 + n;
#pragma unroll
    for (int i = 0; i < 2; ++i) {
#pragma unroll
      for (int r = 0; r < 8; ++r) {
        const int qg = qrow0[i] + r;
        float v0 = s[i][0][r] * scale;
        float v1 = s[i][1][r] * scale;
        if (kcol0 > qg)      v0 = -1e30f;
        if (kcol0 + 16 > qg) v1 = -1e30f;

        float mx = fmaxf(v0, v1);
        mx = fmaxf(mx, __shfl_xor(mx, 1, 32));
        mx = fmaxf(mx, __shfl_xor(mx, 2, 32));
        mx = fmaxf(mx, __shfl_xor(mx, 4, 32));
        mx = fmaxf(mx, __shfl_xor(mx, 8, 32));
        const float mnew  = fmaxf(mrow[i][r], mx);
        const float alpha = __expf(mrow[i][r] - mnew);
        const float p0 = __expf(v0 - mnew);
        const float p1 = __expf(v1 - mnew);
        float rs = p0 + p1;
        rs += __shfl_xor(rs, 1, 32);
        rs += __shfl_xor(rs, 2, 32);
        rs += __shfl_xor(rs, 4, 32);
        rs += __shfl_xor(rs, 8, 32);
        lrow[i][r] = lrow[i][r] * alpha + rs;
        mrow[i][r] = mnew;
#pragma unroll
        for (int j = 0; j < 4; ++j) acc[i][j][r] *= alpha;
        s[i][0][r] = p0;
        s[i][1][r] = p1;
      }
      // C-layout -> A-layout transpose through LDS (16x32 bf16 per tile)
      __bf16* pb = pbuf[wslot] + i * 16 * 32;
#pragma unroll
      for (int r = 0; r < 8; ++r) {
        const int row = half * 8 + r;
        pb[row * 32 + n]      = (__bf16)s[i][0][r];
        pb[row * 32 + 16 + n] = (__bf16)s[i][1][r];
      }
    }

    const bf16x16 pa0 = load_fragA(pbuf[wslot],           32, lane);
    const bf16x16 pa1 = load_fragA(pbuf[wslot] + 16 * 32, 32, lane);

    // ---- acc += P @ V (shared V fragments; V stored [h][d][t], contiguous in k) ----
    const __bf16* vb = vhead + (size_t)(kb * 32);
    if (kb < qt) {
      __builtin_prefetch(vb + 32 + (size_t)n * 4 * T_SEQ, 0, 1);
    }
#pragma unroll
    for (int j = 0; j < 4; ++j) {
      const bf16x16 V = load_fragB(vb + (size_t)(j * 16) * T_SEQ, T_SEQ, lane);
      acc[0][j] = wmma_bf16(pa0, V, acc[0][j]);
      acc[1][j] = wmma_bf16(pa1, V, acc[1][j]);
    }
  }

  // ---- epilogue: y = acc / l, bf16 [T][C] for the projection GEMM ----
#pragma unroll
  for (int i = 0; i < 2; ++i) {
#pragma unroll
    for (int r = 0; r < 8; ++r) {
      const float inv = 1.0f / lrow[i][r];
      const size_t base = (size_t)(qrow0[i] + r) * C_DIM + h * HD + n;
      yb[base]      = (__bf16)(acc[i][0][r] * inv);
      yb[base + 16] = (__bf16)(acc[i][1][r] * inv);
      yb[base + 32] = (__bf16)(acc[i][2][r] * inv);
      yb[base + 48] = (__bf16)(acc[i][3][r] * inv);
    }
  }
}

// ---------------- output projection: 32x64 tile per wave, f32 out ----------------

__global__ __launch_bounds__(256) void proj_gemm_kernel(const __bf16* __restrict__ yb,
                                                        const __bf16* __restrict__ wpT,
                                                        float* __restrict__ out) {
  const int lane = threadIdx.x & 31;
  const int wave = blockIdx.x * 8 + (threadIdx.x >> 5);
  const int tiles_n = C_DIM / 64;                    // 12
  if (wave >= (T_SEQ / 32) * tiles_n) return;
  const int tm = wave / tiles_n;
  const int tn = wave % tiles_n;

  const __bf16* a0 = yb  + (size_t)(tm * 32)      * C_DIM;
  const __bf16* a1 = yb  + (size_t)(tm * 32 + 16) * C_DIM;
  const __bf16* bt = wpT + (size_t)(tn * 64)      * C_DIM;

  f32x8 acc[2][4];
#pragma unroll
  for (int i = 0; i < 2; ++i)
#pragma unroll
    for (int j = 0; j < 4; ++j) acc[i][j] = fzero();

#pragma unroll 2
  for (int kk = 0; kk < C_DIM; kk += 32) {
    const bf16x16 A0 = load_fragA(a0 + kk, C_DIM, lane);
    const bf16x16 A1 = load_fragA(a1 + kk, C_DIM, lane);
#pragma unroll
    for (int j = 0; j < 4; ++j) {
      const bf16x16 B = load_fragB(bt + (size_t)(j * 16) * C_DIM + kk, C_DIM, lane);
      acc[0][j] = wmma_bf16(A0, B, acc[0][j]);
      acc[1][j] = wmma_bf16(A1, B, acc[1][j]);
    }
  }

  const int half = lane >> 4;
  const int n    = lane & 15;
#pragma unroll
  for (int i = 0; i < 2; ++i) {
#pragma unroll
    for (int r = 0; r < 8; ++r) {
      const int row = tm * 32 + i * 16 + half * 8 + r;
#pragma unroll
      for (int j = 0; j < 4; ++j)
        out[(size_t)row * C_DIM + tn * 64 + j * 16 + n] = acc[i][j][r];
    }
  }
}

// ---------------- host launch ----------------

extern "C" void kernel_launch(void* const* d_in, const int* in_sizes, int n_in,
                              void* d_out, int out_size, void* d_ws, size_t ws_size,
                              hipStream_t stream) {
  (void)in_sizes; (void)n_in; (void)out_size; (void)ws_size;
  const float* x      = (const float*)d_in[0];   // [T, C]
  const float* w_attn = (const float*)d_in[1];   // [C, 3C]
  const float* w_proj = (const float*)d_in[2];   // [C, C]
  float* out = (float*)d_out;                    // [T, C]

  char* ws = (char*)d_ws;
  size_t off = 0;
  __bf16* xb  = (__bf16*)(ws + off); off += (size_t)T_SEQ * C_DIM * 2;   // 6.0 MB
  __bf16* waT = (__bf16*)(ws + off); off += (size_t)C3 * C_DIM * 2;      // 3.4 MB
  __bf16* wpT = (__bf16*)(ws + off); off += (size_t)C_DIM * C_DIM * 2;   // 1.1 MB
  __bf16* qh  = (__bf16*)(ws + off); off += (size_t)H_NUM * T_SEQ * HD * 2;
  __bf16* kh  = (__bf16*)(ws + off); off += (size_t)H_NUM * T_SEQ * HD * 2;
  __bf16* vTh = (__bf16*)(ws + off); off += (size_t)H_NUM * HD * T_SEQ * 2;
  __bf16* yb  = (__bf16*)(ws + off); off += (size_t)T_SEQ * C_DIM * 2;

  const int nx = T_SEQ * C_DIM;
  cvt_f32_bf16<<<(nx + 255) / 256, 256, 0, stream>>>(x, xb, nx);

  const int nwa = C_DIM * C3;
  cvt_transpose_bf16<<<(nwa + 255) / 256, 256, 0, stream>>>(w_attn, waT, C_DIM, C3);

  const int nwp = C_DIM * C_DIM;
  cvt_transpose_bf16<<<(nwp + 255) / 256, 256, 0, stream>>>(w_proj, wpT, C_DIM, C_DIM);

  const int qkv_waves = (T_SEQ / 32) * (C3 / 64);                 // 4608
  qkv_gemm_kernel<<<qkv_waves / 8, 256, 0, stream>>>(xb, waT, qh, kh, vTh);

  const int attn_waves = H_NUM * (T_SEQ / 32);                    // 1536
  attn_kernel<<<attn_waves / 8, 256, 0, stream>>>(qh, kh, vTh, yb);

  const int proj_waves = (T_SEQ / 32) * (C_DIM / 64);             // 1536
  proj_gemm_kernel<<<proj_waves / 8, 256, 0, stream>>>(yb, wpT, out);
}
